// CustomAttention_80367428043050
// MI455X (gfx1250) — matrix operand order repaired and checked by reading, not compiled, not run on hardware
//
#include <hip/hip_runtime.h>

// ---------------------------------------------------------------------------
// MHA for B=2,S=2048,E=1024,H=16,D=64 on gfx1250 (wave32, WMMA 16x16x32 bf16).
// Compute-bound (~69 GFLOP vs ~6us of HBM traffic at 23.3 TB/s), so every
// matmul goes through v_wmma_f32_16x16x32_bf16 with fp32 accumulation.
// bf16->bf16 tile staging uses GLOBAL_LOAD_ASYNC_TO_LDS_B128 (ASYNCcnt path).
// ---------------------------------------------------------------------------

typedef unsigned short u16;
typedef unsigned int   u32;

typedef __attribute__((ext_vector_type(16))) __bf16 v16bf;
typedef __attribute__((ext_vector_type(8)))  float  v8f;
typedef __attribute__((ext_vector_type(4)))  int    v4i;

#define BB 2
#define SS 2048
#define EE 1024
#define HH 16
#define DD 64
#define MM (BB * SS)   // 4096 rows for all GEMMs

#if __has_builtin(__builtin_amdgcn_global_load_async_to_lds_b128)
#define HAVE_ASYNC_LDS 1
typedef __attribute__((address_space(1))) v4i* gv4i_p;   // global 128-bit chunk
typedef __attribute__((address_space(3))) v4i* lv4i_p;   // LDS 128-bit chunk
#endif

// 16-byte global -> LDS copy. Async (ASYNCcnt) when the toolchain exposes the
// gfx1250 builtin; plain VGPR round-trip otherwise.
__device__ __forceinline__ void copy16_g2l(const void* g, void* l) {
#ifdef HAVE_ASYNC_LDS
  __builtin_amdgcn_global_load_async_to_lds_b128((gv4i_p)(g), (lv4i_p)(l), 0, 0);
#else
  *(uint4*)l = *(const uint4*)g;
#endif
}
__device__ __forceinline__ void async_wait() {
#ifdef HAVE_ASYNC_LDS
#if __has_builtin(__builtin_amdgcn_s_wait_asynccnt)
  __builtin_amdgcn_s_wait_asynccnt(0);
#else
  asm volatile("s_wait_asynccnt 0x0" ::: "memory");
#endif
#endif
}

union FragU { u32 u[8]; v16bf v; };

__device__ __forceinline__ u16 f2bf(float f) {
  union { float f; u32 u; } x; x.f = f;
  u32 r = x.u + 0x7fffu + ((x.u >> 16) & 1u);   // round-to-nearest-even
  return (u16)(r >> 16);
}
__device__ __forceinline__ u32 pack2(float a, float b) {
  return ((u32)f2bf(b) << 16) | (u32)f2bf(a);
}

// A fragment (16x32 MxK, 16-bit): lanes 0-15 hold row M=lane, K in {0..7,16..23};
// lanes 16-31 hold row M=lane-16, K in {8..15,24..31}. Row stride must be a
// multiple of 8 u16 so all loads are 16B aligned -> ds_load_b128.
__device__ __forceinline__ v16bf load_a_frag(const u16* base, int stride, int lane) {
  const int row = lane & 15;
  const int kb  = (lane >> 4) << 3;             // 0 or 8
  const u16* p = base + row * stride + kb;
  uint4 lo = *(const uint4*)(p);
  uint4 hi = *(const uint4*)(p + 16);
  FragU f;
  f.u[0] = lo.x; f.u[1] = lo.y; f.u[2] = lo.z; f.u[3] = lo.w;
  f.u[4] = hi.x; f.u[5] = hi.y; f.u[6] = hi.z; f.u[7] = hi.w;
  return f.v;
}

// B fragment (32x16 KxN, 16-bit): lane gives N=lane&15; lanes 0-15 hold K in {0..7,16..23},
// lanes 16-31 hold K in {8..15,24..31}, two K packed per VGPR. LDS layout is [n][k].
__device__ __forceinline__ v16bf load_b_frag(const u16* base, int stride, int lane) {
  const int n  = lane & 15;
  const int kb = (lane >> 4) << 3;              // 0 or 8
  const u16* p = base + n * stride + kb;
  uint4 lo = *(const uint4*)(p);
  uint4 hi = *(const uint4*)(p + 16);
  FragU f;
  f.u[0] = lo.x; f.u[1] = lo.y; f.u[2] = lo.z; f.u[3] = lo.w;
  f.u[4] = hi.x; f.u[5] = hi.y; f.u[6] = hi.z; f.u[7] = hi.w;
  return f.v;
}

__device__ __forceinline__ v8f wmma_bf16(v16bf a, v16bf b, v8f c) {
  // (neg_a, A, neg_b, B, c_mod, C, reuse_a, reuse_b)
  return __builtin_amdgcn_wmma_f32_16x16x32_bf16(false, a, false, b, (short)0, c,
                                                 false, false);
}

// ---------------------------------------------------------------------------
// Y[m,n] = sum_k A[m,k] * W[n,k] + bias[n]   (torch Linear: x @ W.T + b)
// M=4096, N=K=1024. WG tile 128x64, TK=64, 8 waves -> each wave 32x32 (2x2 WMMA).
// ---------------------------------------------------------------------------
template <bool ABF, bool OBF>
__global__ __launch_bounds__(256) void gemm_bias_kernel(
    const void* __restrict__ A_, const float* __restrict__ W,
    const float* __restrict__ bias, void* __restrict__ Y_) {
  constexpr int TM = 128, TN = 64, TK = 64, LDT = TK + 8;  // 72 u16 = 144B (16B mult)
  constexpr int Kd = EE, Nd = EE;
  __shared__ u16 As[TM][LDT];
  __shared__ u16 Bs[TN][LDT];

  const int tid = threadIdx.x, lane = tid & 31, w = tid >> 5;
  const int m0 = blockIdx.y * TM, n0 = blockIdx.x * TN;
  const int wm = (w & 3) * 32, wn = (w >> 2) * 32;
  const float* Af = (const float*)A_;
  const u16*   Ab = (const u16*)A_;

  v8f acc[2][2] = {};

  for (int k0 = 0; k0 < Kd; k0 += TK) {
    // Stage A tile.
    if constexpr (ABF) {
      // Already bf16: DMA 16B chunks straight into LDS (ASYNCcnt path).
      for (int idx = tid; idx < TM * 8; idx += 256) {
        int row = idx >> 3, c = idx & 7;                // 8x 16B chunks per row
        copy16_g2l(&Ab[(size_t)(m0 + row) * Kd + k0 + c * 8], &As[row][c * 8]);
      }
    } else {
      // fp32 -> bf16 convert through VGPRs.
      for (int idx = tid; idx < TM * TK / 2; idx += 256) {
        int row = idx >> 5, kp = idx & 31;              // TK/2 == 32
        int gm = m0 + row, gk = k0 + kp * 2;
        float2 f = *(const float2*)(Af + (size_t)gm * Kd + gk);
        *(u32*)&As[row][kp * 2] = pack2(f.x, f.y);
        if (k0 + TK < Kd)
          __builtin_prefetch(Af + (size_t)gm * Kd + gk + TK, 0, 0); // global_prefetch_b8
      }
    }
    // Stage W tile, layout [n][k] which is exactly W's row-major storage.
    for (int idx = tid; idx < TN * TK / 2; idx += 256) {
      int row = idx >> 5, kp = idx & 31;
      float2 f = *(const float2*)(W + (size_t)(n0 + row) * Kd + k0 + kp * 2);
      *(u32*)&Bs[row][kp * 2] = pack2(f.x, f.y);
    }
    if constexpr (ABF) async_wait();
    __syncthreads();

#pragma unroll
    for (int kk = 0; kk < TK; kk += 32) {
      v16bf a0 = load_a_frag(&As[wm][kk],      LDT, lane);
      v16bf a1 = load_a_frag(&As[wm + 16][kk], LDT, lane);
      v16bf b0 = load_b_frag(&Bs[wn][kk],      LDT, lane);
      v16bf b1 = load_b_frag(&Bs[wn + 16][kk], LDT, lane);
      acc[0][0] = wmma_bf16(a0, b0, acc[0][0]);
      acc[0][1] = wmma_bf16(a0, b1, acc[0][1]);
      acc[1][0] = wmma_bf16(a1, b0, acc[1][0]);
      acc[1][1] = wmma_bf16(a1, b1, acc[1][1]);
    }
    __syncthreads();
  }

  // C/D layout: lane l, VGPR r -> M = r + 8*(l>=16), N = l&15.
  const int lane15 = lane & 15, hi8 = (lane >> 4) << 3;
  float* Yf = (float*)Y_;
  u16*   Yb = (u16*)Y_;
#pragma unroll
  for (int mi = 0; mi < 2; ++mi)
#pragma unroll
    for (int ni = 0; ni < 2; ++ni) {
      int gn = n0 + wn + ni * 16 + lane15;
      float bv = bias[gn];
#pragma unroll
      for (int r = 0; r < 8; ++r) {
        int gm = m0 + wm + mi * 16 + hi8 + r;
        float v = acc[mi][ni][r] + bv;
        if (OBF) Yb[(size_t)gm * Nd + gn] = f2bf(v);
        else     Yf[(size_t)gm * Nd + gn] = v;
      }
    }
}

// ---------------------------------------------------------------------------
// Flash attention: grid (S/128, H, B), 256 threads = 8 waves, wave w owns
// Q rows [qt*128 + 16w, +16). Online softmax, P round-trips through
// wave-private LDS to convert C-layout -> A-layout for P@V.
// Q and K tiles are staged with async global->LDS DMA; V goes through VGPRs
// because it must land transposed ([d][kv]) for the P@V B-fragments.
// ---------------------------------------------------------------------------
__global__ __launch_bounds__(256) void flash_attn_kernel(
    const u16* __restrict__ Q, const u16* __restrict__ Kh,
    const u16* __restrict__ V, u16* __restrict__ O,
    const int* __restrict__ causal_p) {
  constexpr int LDQ = DD + 8;   // 72 u16 = 144B, multiple of 16B
  constexpr int LDV = 64 + 8;   // 72
  __shared__ u16 Qs[128][LDQ];          // [q row][d]
  __shared__ u16 Ks[64][LDQ];           // [kv row][d] == B layout for Q@K^T
  __shared__ u16 Vt[64][LDV];           // [d][kv row] == B layout for P@V
  __shared__ u16 Ps[8][16][LDV];        // per-wave P (A layout source)

  const int qt = blockIdx.x, h = blockIdx.y, b = blockIdx.z;
  const int tid = threadIdx.x, lane = tid & 31, w = tid >> 5;
  const int lane15 = lane & 15, hi8 = (lane >> 4) << 3;
  const int q0 = qt * 128;

  // Q tile: 128 rows x 128B -> 1024 async 16B chunks.
  for (int idx = tid; idx < 128 * 8; idx += 256) {
    int row = idx >> 3, c = idx & 7;
    copy16_g2l(&Q[((size_t)b * SS + q0 + row) * EE + h * DD + c * 8],
               &Qs[row][c * 8]);
  }

  const int causal = *causal_p;
  const int njt = causal ? (qt * 2 + 2) : (SS / 64);
  const u32* Vg = (const u32*)V;

  v8f o[4] = {};
  float mrow[8], lrow[8];
#pragma unroll
  for (int r = 0; r < 8; ++r) { mrow[r] = -1e30f; lrow[r] = 0.f; }

  for (int j = 0; j < njt; ++j) {
    const int kv0 = j * 64;
    __syncthreads();   // previous tile's compute done reading Ks/Vt
    // K tile: async DMA, 64 rows x 8 chunks.
    for (int idx = tid; idx < 64 * 8; idx += 256) {
      int row = idx >> 3, c = idx & 7;
      copy16_g2l(&Kh[((size_t)b * SS + kv0 + row) * EE + h * DD + c * 8],
                 &Ks[row][c * 8]);
    }
    // V tile: through VGPRs, stored transposed into [d][kv].
    for (int idx = tid; idx < 64 * 32; idx += 256) {
      int row = idx >> 5, kp = idx & 31;
      u32 v2 = Vg[(((size_t)b * SS + kv0 + row) * EE + h * DD) / 2 + kp];
      Vt[kp * 2][row]     = (u16)(v2 & 0xffffu);
      Vt[kp * 2 + 1][row] = (u16)(v2 >> 16);
    }
    async_wait();      // this wave's async LDS writes (Q on j==0, K always)
    __syncthreads();   // all waves' tiles visible

    // S = Q K^T  (16x64 per wave -> 4 N-fragments, K=64 in 2 steps)
    v8f s[4] = {};
#pragma unroll
    for (int kk = 0; kk < 64; kk += 32) {
      v16bf aq = load_a_frag(&Qs[w * 16][kk], LDQ, lane);
#pragma unroll
      for (int ni = 0; ni < 4; ++ni)
        s[ni] = wmma_bf16(aq, load_b_frag(&Ks[ni * 16][kk], LDQ, lane), s[ni]);
    }

    // Online softmax; each row lives across one 16-lane half -> shfl_xor<16.
#pragma unroll
    for (int r = 0; r < 8; ++r) {
      const int qrow = q0 + w * 16 + hi8 + r;
      float xm = -1e30f;
#pragma unroll
      for (int ni = 0; ni < 4; ++ni) {
        float x = s[ni][r] * 0.125f;        // D^-0.5
        int col = kv0 + ni * 16 + lane15;
        if (causal && col > qrow) x = -1e30f;
        s[ni][r] = x;
        xm = fmaxf(xm, x);
      }
#pragma unroll
      for (int msk = 1; msk < 16; msk <<= 1) xm = fmaxf(xm, __shfl_xor(xm, msk, 32));
      float mn = fmaxf(mrow[r], xm);
      float alpha = __expf(mrow[r] - mn);
      float rs = 0.f;
#pragma unroll
      for (int ni = 0; ni < 4; ++ni) {
        float p = __expf(s[ni][r] - mn);
        s[ni][r] = p;
        rs += p;
      }
#pragma unroll
      for (int msk = 1; msk < 16; msk <<= 1) rs += __shfl_xor(rs, msk, 32);
      lrow[r] = lrow[r] * alpha + rs;
      mrow[r] = mn;
#pragma unroll
      for (int di = 0; di < 4; ++di) o[di][r] *= alpha;
#pragma unroll
      for (int ni = 0; ni < 4; ++ni)       // C-layout -> LDS (A-layout source)
        Ps[w][hi8 + r][ni * 16 + lane15] = f2bf(s[ni][r]);
    }

    // O += P V   (wave-private Ps; hw DScnt ordering, no barrier needed)
#pragma unroll
    for (int kk = 0; kk < 64; kk += 32) {
      v16bf ap = load_a_frag(&Ps[w][0][kk], LDV, lane);
#pragma unroll
      for (int di = 0; di < 4; ++di)
        o[di] = wmma_bf16(ap, load_b_frag(&Vt[di * 16][kk], LDV, lane), o[di]);
    }
  }

  // Normalize and write bf16 [B,S,E] (head columns interleaved).
#pragma unroll
  for (int r = 0; r < 8; ++r) {
    float inv = 1.f / lrow[r];
    int gm = q0 + w * 16 + hi8 + r;
#pragma unroll
    for (int di = 0; di < 4; ++di)
      O[((size_t)b * SS + gm) * EE + h * DD + di * 16 + lane15] =
          f2bf(o[di][r] * inv);
  }
}

// ---------------------------------------------------------------------------
extern "C" void kernel_launch(void* const* d_in, const int* in_sizes, int n_in,
                              void* d_out, int out_size, void* d_ws, size_t ws_size,
                              hipStream_t stream) {
  (void)in_sizes; (void)n_in; (void)out_size; (void)ws_size;
  const float* query = (const float*)d_in[0];
  const float* key   = (const float*)d_in[1];
  const float* value = (const float*)d_in[2];
  const float* Wq = (const float*)d_in[3];  const float* bq = (const float*)d_in[4];
  const float* Wk = (const float*)d_in[5];  const float* bk = (const float*)d_in[6];
  const float* Wv = (const float*)d_in[7];  const float* bv = (const float*)d_in[8];
  const float* Wo = (const float*)d_in[9];  const float* bo = (const float*)d_in[10];
  const int* is_causal = (const int*)d_in[11];

  // Workspace: 4 bf16 [B,S,E] buffers = 4 * 8MB = 32MB.
  const size_t per = (size_t)MM * EE;
  u16* Qb = (u16*)d_ws;
  u16* Kb = Qb + per;
  u16* Vb = Kb + per;
  u16* Ab = Vb + per;

  dim3 gblk(256);
  dim3 ggemm(EE / 64, MM / 128);          // (16, 32)
  gemm_bias_kernel<false, true><<<ggemm, gblk, 0, stream>>>(query, Wq, bq, Qb);
  gemm_bias_kernel<false, true><<<ggemm, gblk, 0, stream>>>(key,   Wk, bk, Kb);
  gemm_bias_kernel<false, true><<<ggemm, gblk, 0, stream>>>(value, Wv, bv, Vb);

  dim3 gattn(SS / 128, HH, BB);           // (16, 16, 2)
  flash_attn_kernel<<<gattn, gblk, 0, stream>>>(Qb, Kb, Vb, Ab, is_causal);

  gemm_bias_kernel<true, false><<<ggemm, gblk, 0, stream>>>(Ab, Wo, bo, d_out);
}
